// _HierarchicalGatedPool_48455821033935
// MI455X (gfx1250) — matrix operand order repaired and checked by reading, not compiled, not run on hardware
//
#include <hip/hip_runtime.h>
#include <hip/hip_bf16.h>

// ---------------- problem constants (match reference) ----------------
#define N_ATOMS 262144
#define NB      2048
#define HIDD    256
#define CNDD    128
#define KTOT    384        // HID + CND
#define KCL     16         // n_clusters
#define PH      8          // pool heads
#define DHD     32         // head dim

// ---------------- CDNA5 WMMA types ----------------
typedef __attribute__((ext_vector_type(16))) __bf16 v16bf;
typedef __attribute__((ext_vector_type(8)))  float  v8f;

union Frag {
    v16bf v;
    __bf16 el[16];
    uint4  q[2];
};

__device__ __forceinline__ v8f wmma_bf16(const Frag& a, const Frag& b, v8f c) {
    // D = A(16x32 bf16) x B(32x16 bf16) + C(16x16 f32)
    return __builtin_amdgcn_wmma_f32_16x16x32_bf16(
        /*neg_a=*/false, a.v, /*neg_b=*/false, b.v,
        /*c_mod=*/(short)0, c, /*reuse_a=*/false, /*reuse_b=*/false);
}

__device__ __forceinline__ float gelu_erf(float x) {
    return 0.5f * x * (1.0f + erff(x * 0.70710678118654752f));
}

// ---------------- small utility kernels ----------------
__global__ void k_f2bf(const float* __restrict__ src, __bf16* __restrict__ dst, int n) {
    int i = blockIdx.x * 256 + threadIdx.x;
    if (i < n) dst[i] = (__bf16)src[i];
}

__global__ void k_starts_init(int* __restrict__ starts) {
    int i = blockIdx.x * 256 + threadIdx.x;
    if (i <= NB) starts[i] = N_ATOMS;
}

// batch is sorted; starts[b] = first atom index of graph b, starts[NB] = N.
__global__ void k_starts_fill(const int* __restrict__ batch, int* __restrict__ starts) {
    int n = blockIdx.x * 256 + threadIdx.x;
    if (n >= N_ATOMS) return;
    int cur  = batch[n];
    int prev = (n == 0) ? -1 : batch[n - 1];
    for (int b = prev + 1; b <= cur; ++b) starts[b] = n;
}

// ---------------- Stage 1: gate GEMM + sigmoid gate + h_gated ----------------
// Block tile: 32 rows x 256 cols, K = 384 in chunks of 32.
// 8 waves: 2 (M) x 4 (N); per-wave 16 rows x 64 cols = 4 WMMA accumulators.
__global__ __launch_bounds__(256) void k_gate(
    const float* __restrict__ h_atom, const float* __restrict__ c_atom,
    const __bf16* __restrict__ wW1, const float* __restrict__ b1,
    const float* __restrict__ W2, const float* __restrict__ b2,
    __bf16* __restrict__ h_gated)
{
    __shared__ __align__(16) __bf16 At[32][32];
    __shared__ __align__(16) __bf16 Bt[32][HIDD];
    __shared__ __align__(16) float  Gt[32][HIDD];
    __shared__ float aRow[32];

    const int t = threadIdx.x;
    const int wave = t >> 5, lane = t & 31;
    const int wm = wave >> 2, wn = wave & 3;
    const int row0 = blockIdx.x * 32;

    v8f acc0 = {}, acc1 = {}, acc2 = {}, acc3 = {};

    for (int kc = 0; kc < KTOT; kc += 32) {
        // ---- stage A tile (fp32 -> bf16): 32 rows x 32 k
        #pragma unroll
        for (int i = 0; i < 4; ++i) {
            int idx = t + i * 256;
            int r = idx >> 5, c = idx & 31;
            int k = kc + c;
            float x = (k < HIDD)
                    ? h_atom[(size_t)(row0 + r) * HIDD + k]
                    : c_atom[(size_t)(row0 + r) * CNDD + (k - HIDD)];
            At[r][c] = (__bf16)x;
        }
        // ---- stage B tile (bf16 weights): 32 k x 256 n (uint4 = 8 bf16)
        #pragma unroll
        for (int i = 0; i < 4; ++i) {
            int chunk = t + i * 256;              // 1024 chunks
            int r = chunk >> 5;                   // 32 chunks per row
            int c = (chunk & 31) * 8;
            *(uint4*)&Bt[r][c] = *(const uint4*)&wW1[(size_t)(kc + r) * HIDD + c];
        }
        // prefetch next weight K-slab into caches
        if (kc + 32 < KTOT && t < 32)
            __builtin_prefetch(&wW1[(size_t)(kc + 32 + t) * HIDD], 0, 0);
        __syncthreads();

        // ---- A fragment: lanes 0-15 rows M, K {0..7,16..23}; lanes 16-31 K {8..15,24..31}
        Frag a;
        {
            int m  = wm * 16 + (lane & 15);
            int kl = (lane < 16) ? 0 : 8;
            a.q[0] = *(const uint4*)&At[m][kl];
            a.q[1] = *(const uint4*)&At[m][kl + 16];
        }
        // ---- B fragments: lane%16 = N col; K {0..7,16..23} / {8..15,24..31} per lane half
        #pragma unroll
        for (int j = 0; j < 4; ++j) {
            Frag b;
            int n0 = wn * 64 + j * 16;
            int nn = n0 + (lane & 15), kb = (lane < 16) ? 0 : 8;
            #pragma unroll
            for (int e = 0; e < 8; ++e) { b.el[e] = Bt[kb + e][nn]; b.el[8 + e] = Bt[kb + 16 + e][nn]; }
            v8f& acc = (j == 0) ? acc0 : (j == 1) ? acc1 : (j == 2) ? acc2 : acc3;
            acc = wmma_bf16(a, b, acc);
        }
        __syncthreads();
    }

    // ---- epilogue: bias + GELU -> Gt (fp32)
    #pragma unroll
    for (int j = 0; j < 4; ++j) {
        v8f acc = (j == 0) ? acc0 : (j == 1) ? acc1 : (j == 2) ? acc2 : acc3;
        int col   = wn * 64 + j * 16 + (lane & 15);
        int rbase = wm * 16 + ((lane >= 16) ? 8 : 0);
        #pragma unroll
        for (int r = 0; r < 8; ++r)
            Gt[rbase + r][col] = gelu_erf(acc[r] + b1[col]);
    }
    __syncthreads();

    // ---- alpha = sigmoid(g . W2 + b2): 8 lanes per row, shuffle reduce
    {
        int r = t >> 3, seg = t & 7;
        float p = 0.f;
        #pragma unroll 4
        for (int c = seg * 32; c < seg * 32 + 32; ++c) p += Gt[r][c] * W2[c];
        p += __shfl_down(p, 4, 8);
        p += __shfl_down(p, 2, 8);
        p += __shfl_down(p, 1, 8);
        if (seg == 0) aRow[r] = 1.f / (1.f + __expf(-(p + b2[0])));
    }
    __syncthreads();

    // ---- h_gated = alpha * h_atom (bf16 out)
    #pragma unroll
    for (int i = 0; i < 32; ++i) {
        int idx = t + i * 256;
        int r = idx >> 8, c = idx & 255;
        float hg = aRow[r] * h_atom[(size_t)(row0 + r) * HIDD + c];
        h_gated[(size_t)(row0 + r) * HIDD + c] = (__bf16)hg;
    }
}

// ---------------- Stage 2a: assign GEMM + GELU + @A2 + softmax -> S ----------------
__global__ __launch_bounds__(256) void k_assign(
    const __bf16* __restrict__ h_gated,
    const __bf16* __restrict__ wA1, const float* __restrict__ ab1,
    const float* __restrict__ A2, const float* __restrict__ ab2,
    __bf16* __restrict__ S)
{
    __shared__ __align__(16) __bf16 At[32][32];
    __shared__ __align__(16) __bf16 Bt[32][HIDD];
    __shared__ __align__(16) float  Tt[32][HIDD];
    __shared__ float Lg[32][KCL];

    const int t = threadIdx.x;
    const int wave = t >> 5, lane = t & 31;
    const int wm = wave >> 2, wn = wave & 3;
    const int row0 = blockIdx.x * 32;

    v8f acc0 = {}, acc1 = {}, acc2 = {}, acc3 = {};

    for (int kc = 0; kc < HIDD; kc += 32) {
        if (t < 128) {  // A tile: 32x32 bf16 = 128 uint4
            int r = t >> 2, c = (t & 3) * 8;
            *(uint4*)&At[r][c] = *(const uint4*)&h_gated[(size_t)(row0 + r) * HIDD + kc + c];
        }
        #pragma unroll
        for (int i = 0; i < 4; ++i) {
            int chunk = t + i * 256;
            int r = chunk >> 5;
            int c = (chunk & 31) * 8;
            *(uint4*)&Bt[r][c] = *(const uint4*)&wA1[(size_t)(kc + r) * HIDD + c];
        }
        if (kc + 32 < HIDD && t < 32)
            __builtin_prefetch(&wA1[(size_t)(kc + 32 + t) * HIDD], 0, 0);
        __syncthreads();

        Frag a;
        {
            int m  = wm * 16 + (lane & 15);
            int kl = (lane < 16) ? 0 : 8;
            a.q[0] = *(const uint4*)&At[m][kl];
            a.q[1] = *(const uint4*)&At[m][kl + 16];
        }
        #pragma unroll
        for (int j = 0; j < 4; ++j) {
            Frag b;
            int n0 = wn * 64 + j * 16;
            int nn = n0 + (lane & 15), kb = (lane < 16) ? 0 : 8;
            #pragma unroll
            for (int e = 0; e < 8; ++e) { b.el[e] = Bt[kb + e][nn]; b.el[8 + e] = Bt[kb + 16 + e][nn]; }
            v8f& acc = (j == 0) ? acc0 : (j == 1) ? acc1 : (j == 2) ? acc2 : acc3;
            acc = wmma_bf16(a, b, acc);
        }
        __syncthreads();
    }

    #pragma unroll
    for (int j = 0; j < 4; ++j) {
        v8f acc = (j == 0) ? acc0 : (j == 1) ? acc1 : (j == 2) ? acc2 : acc3;
        int col   = wn * 64 + j * 16 + (lane & 15);
        int rbase = wm * 16 + ((lane >= 16) ? 8 : 0);
        #pragma unroll
        for (int r = 0; r < 8; ++r)
            Tt[rbase + r][col] = gelu_erf(acc[r] + ab1[col]);
    }
    __syncthreads();

    // logits = Tt @ A2 + ab2 : 32 rows x 16 clusters, 2 tasks/thread
    #pragma unroll
    for (int i = 0; i < 2; ++i) {
        int task = t + i * 256;
        int r = task >> 4, k = task & 15;
        float s = ab2[k];
        #pragma unroll 4
        for (int c = 0; c < HIDD; ++c) s += Tt[r][c] * A2[c * KCL + k];
        Lg[r][k] = s;
    }
    __syncthreads();

    if (t < 32) {
        int r = t;
        float mx = -1e30f;
        #pragma unroll
        for (int k = 0; k < KCL; ++k) mx = fmaxf(mx, Lg[r][k]);
        float e[KCL], se = 0.f;
        #pragma unroll
        for (int k = 0; k < KCL; ++k) { e[k] = __expf(Lg[r][k] - mx); se += e[k]; }
        float inv = 1.f / se;
        #pragma unroll
        for (int k = 0; k < KCL; ++k)
            S[(size_t)(row0 + r) * KCL + k] = (__bf16)(e[k] * inv);
    }
}

// ---------------- Stage 2b: ragged segment GEMM -> cluster_raw ----------------
// One wave per (graph, 16-col tile): C[16x16] += S_chunk^T(16x32) x h_gated_chunk(32x16)
__global__ __launch_bounds__(256) void k_cluster(
    const __bf16* __restrict__ S, const __bf16* __restrict__ h_gated,
    const int* __restrict__ starts, __bf16* __restrict__ cluster_raw)
{
    __shared__ __align__(16) __bf16 Sc[8][32][KCL];

    const int t = threadIdx.x, wave = t >> 5, lane = t & 31;
    const int task = blockIdx.x * 8 + wave;          // < NB*16
    const int b = task >> 4, col0 = (task & 15) * 16;
    const int beg = starts[b], end = starts[b + 1];

    v8f acc = {};

    for (int n0 = beg; n0 < end; n0 += 32) {
        const int n = n0 + lane;
        // stage 32 S rows (zero-padded tail) for the transposed A fragment
        if (n < end) {
            const uint4* sp = (const uint4*)&S[(size_t)n * KCL];
            *(uint4*)&Sc[wave][lane][0] = sp[0];
            *(uint4*)&Sc[wave][lane][8] = sp[1];
        } else {
            uint4 z = make_uint4(0u, 0u, 0u, 0u);
            *(uint4*)&Sc[wave][lane][0] = z;
            *(uint4*)&Sc[wave][lane][8] = z;
        }
        // A fragment: M = cluster k, K = atom-in-chunk (transpose gather from LDS)
        Frag a;
        {
            int m  = lane & 15;
            int kl = (lane < 16) ? 0 : 8;
            #pragma unroll
            for (int e = 0; e < 8; ++e) {
                a.el[e]     = Sc[wave][kl + e][m];
                a.el[8 + e] = Sc[wave][kl + 16 + e][m];
            }
        }
        // B fragment: N col = col0 + lane%16, K = atom-in-chunk (same order as A)
        Frag bf;
        {
            int nc = col0 + (lane & 15), kb = (lane < 16) ? 0 : 8;
            #pragma unroll
            for (int e = 0; e < 8; ++e) {
                int a0 = n0 + kb + e, a1 = n0 + kb + 16 + e;
                bf.el[e]     = (a0 < end) ? h_gated[(size_t)a0 * HIDD + nc] : (__bf16)0.f;
                bf.el[8 + e] = (a1 < end) ? h_gated[(size_t)a1 * HIDD + nc] : (__bf16)0.f;
            }
        }
        acc = wmma_bf16(a, bf, acc);
    }

    const int kb  = (lane >= 16) ? 8 : 0;
    const int col = col0 + (lane & 15);
    #pragma unroll
    for (int r = 0; r < 8; ++r)
        cluster_raw[((size_t)b * KCL + (kb + r)) * HIDD + col] = (__bf16)acc[r];
}

// ---------------- Stage 3: cluster_proj + K/V + attention + LayerNorm ----------------
// One block per graph; three 16x256x256 WMMA GEMMs, weights L2-resident.
__global__ __launch_bounds__(256) void k_readout(
    const __bf16* __restrict__ cluster_raw,
    const __bf16* __restrict__ wWc, const float* __restrict__ bc,
    const __bf16* __restrict__ wWk, const __bf16* __restrict__ wWv,
    const float* __restrict__ qv, const float* __restrict__ gamma,
    const float* __restrict__ beta, float* __restrict__ out)
{
    __shared__ __align__(16) __bf16 Cl[KCL][HIDD];
    __shared__ __align__(16) __bf16 C2[KCL][HIDD];
    __shared__ float KV[2][KCL][HIDD];
    __shared__ float Lg[PH][KCL];
    __shared__ float Ov[HIDD];
    __shared__ float Red[8][2];
    __shared__ float muS, rsS;

    const int b = blockIdx.x;
    const int t = threadIdx.x, wave = t >> 5, lane = t & 31;

    {   // stage cluster_raw[b] : 16x256 bf16 = 512 uint4
        const uint4* src = (const uint4*)&cluster_raw[(size_t)b * KCL * HIDD];
        uint4* dst = (uint4*)&Cl[0][0];
        dst[t] = src[t];
        dst[t + 256] = src[t + 256];
    }
    __syncthreads();

    // C2 = Cl @ Wc + bc  (2 N-tiles per wave)
    #pragma unroll
    for (int i = 0; i < 2; ++i) {
        int n0 = (wave * 2 + i) * 16;
        v8f acc = {};
        for (int kc = 0; kc < HIDD; kc += 32) {
            Frag a;
            int m  = lane & 15;
            int kl = kc + ((lane < 16) ? 0 : 8);
            a.q[0] = *(const uint4*)&Cl[m][kl];
            a.q[1] = *(const uint4*)&Cl[m][kl + 16];
            Frag bf;
            int nn = n0 + (lane & 15), kb = kc + ((lane < 16) ? 0 : 8);
            #pragma unroll
            for (int e = 0; e < 8; ++e) { bf.el[e] = wWc[(size_t)(kb + e) * HIDD + nn]; bf.el[8 + e] = wWc[(size_t)(kb + 16 + e) * HIDD + nn]; }
            acc = wmma_bf16(a, bf, acc);
        }
        int rb = (lane >= 16) ? 8 : 0, col = n0 + (lane & 15);
        #pragma unroll
        for (int r = 0; r < 8; ++r)
            C2[rb + r][col] = (__bf16)(acc[r] + bc[col]);
    }
    __syncthreads();

    // Kp = C2 @ Wk, Vp = C2 @ Wv : 32 tile-tasks over 8 waves
    #pragma unroll
    for (int i = 0; i < 4; ++i) {
        int task = wave * 4 + i;                  // 0..31
        int mat = task >> 4, n0 = (task & 15) * 16;
        const __bf16* W = mat ? wWv : wWk;
        v8f acc = {};
        for (int kc = 0; kc < HIDD; kc += 32) {
            Frag a;
            int m  = lane & 15;
            int kl = kc + ((lane < 16) ? 0 : 8);
            a.q[0] = *(const uint4*)&C2[m][kl];
            a.q[1] = *(const uint4*)&C2[m][kl + 16];
            Frag bf;
            int nn = n0 + (lane & 15), kb = kc + ((lane < 16) ? 0 : 8);
            #pragma unroll
            for (int e = 0; e < 8; ++e) { bf.el[e] = W[(size_t)(kb + e) * HIDD + nn]; bf.el[8 + e] = W[(size_t)(kb + 16 + e) * HIDD + nn]; }
            acc = wmma_bf16(a, bf, acc);
        }
        int rb = (lane >= 16) ? 8 : 0, col = n0 + (lane & 15);
        #pragma unroll
        for (int r = 0; r < 8; ++r)
            KV[mat][rb + r][col] = acc[r];
    }
    __syncthreads();

    // logits[p,k] = (q[p,:] . Kp[k,p,:]) / sqrt(DH)
    if (t < PH * KCL) {
        int p = t >> 4, k = t & 15;
        float s = 0.f;
        #pragma unroll
        for (int d = 0; d < DHD; ++d) s += qv[p * DHD + d] * KV[0][k][p * DHD + d];
        Lg[p][k] = s * 0.17677669529663687f;      // DH^-0.5
    }
    __syncthreads();
    if (t < PH) {
        float mx = -1e30f;
        #pragma unroll
        for (int k = 0; k < KCL; ++k) mx = fmaxf(mx, Lg[t][k]);
        float se = 0.f;
        #pragma unroll
        for (int k = 0; k < KCL; ++k) { float e = __expf(Lg[t][k] - mx); Lg[t][k] = e; se += e; }
        float inv = 1.f / se;
        #pragma unroll
        for (int k = 0; k < KCL; ++k) Lg[t][k] *= inv;
    }
    __syncthreads();

    // out[col] = sum_k w[p,k] * Vp[k, col]   (p = col/32)
    {
        int p = t >> 5;
        float o = 0.f;
        #pragma unroll
        for (int k = 0; k < KCL; ++k) o += Lg[p][k] * KV[1][k][t];
        Ov[t] = o;
    }
    __syncthreads();

    // LayerNorm over 256
    {
        float o = Ov[t], s = o, s2 = o * o;
        #pragma unroll
        for (int off = 16; off > 0; off >>= 1) {
            s  += __shfl_down(s, off, 32);
            s2 += __shfl_down(s2, off, 32);
        }
        if (lane == 0) { Red[wave][0] = s; Red[wave][1] = s2; }
    }
    __syncthreads();
    if (t == 0) {
        float s = 0.f, s2 = 0.f;
        #pragma unroll
        for (int w = 0; w < 8; ++w) { s += Red[w][0]; s2 += Red[w][1]; }
        float mu = s * (1.f / HIDD);
        float var = s2 * (1.f / HIDD) - mu * mu;
        muS = mu;
        rsS = rsqrtf(var + 1e-5f);
    }
    __syncthreads();
    out[(size_t)b * HIDD + t] = (Ov[t] - muS) * rsS * gamma[t] + beta[t];
}

// ---------------- host-side launcher ----------------
extern "C" void kernel_launch(void* const* d_in, const int* in_sizes, int n_in,
                              void* d_out, int out_size, void* d_ws, size_t ws_size,
                              hipStream_t stream) {
    (void)in_sizes; (void)n_in; (void)out_size; (void)ws_size;
    // input order per setup_inputs():
    const float* h_atom = (const float*)d_in[0];
    const float* c_atom = (const float*)d_in[1];
    const int*   batch  = (const int*)d_in[2];
    const float* W1 = (const float*)d_in[3];
    const float* b1 = (const float*)d_in[4];
    const float* W2 = (const float*)d_in[5];
    const float* b2 = (const float*)d_in[6];
    const float* A1 = (const float*)d_in[7];
    const float* ab1 = (const float*)d_in[8];
    const float* A2 = (const float*)d_in[9];
    const float* ab2 = (const float*)d_in[10];
    const float* Wc = (const float*)d_in[11];
    const float* bc = (const float*)d_in[12];
    const float* qv = (const float*)d_in[13];
    const float* Wk = (const float*)d_in[14];
    const float* Wv = (const float*)d_in[15];
    const float* gamma = (const float*)d_in[16];
    const float* beta  = (const float*)d_in[17];
    float* outp = (float*)d_out;

    // workspace carve-out (256B aligned regions)
    size_t off = 0;
    auto carve = [&](size_t bytes) -> void* {
        void* p = (char*)d_ws + off;
        off += (bytes + 255) & ~(size_t)255;
        return p;
    };
    __bf16* wW1 = (__bf16*)carve((size_t)KTOT * HIDD * 2);
    __bf16* wA1 = (__bf16*)carve((size_t)HIDD * HIDD * 2);
    __bf16* wWc = (__bf16*)carve((size_t)HIDD * HIDD * 2);
    __bf16* wWk = (__bf16*)carve((size_t)HIDD * HIDD * 2);
    __bf16* wWv = (__bf16*)carve((size_t)HIDD * HIDD * 2);
    __bf16* h_gated     = (__bf16*)carve((size_t)N_ATOMS * HIDD * 2);   // 128 MB
    __bf16* Sm          = (__bf16*)carve((size_t)N_ATOMS * KCL * 2);    // 8 MB
    __bf16* cluster_raw = (__bf16*)carve((size_t)NB * KCL * HIDD * 2);  // 16 MB
    int*    starts      = (int*)carve((size_t)(NB + 1) * 4);

    // 0) weight conversion fp32 -> bf16
    k_f2bf<<<(KTOT * HIDD + 255) / 256, 256, 0, stream>>>(W1, wW1, KTOT * HIDD);
    k_f2bf<<<(HIDD * HIDD + 255) / 256, 256, 0, stream>>>(A1, wA1, HIDD * HIDD);
    k_f2bf<<<(HIDD * HIDD + 255) / 256, 256, 0, stream>>>(Wc, wWc, HIDD * HIDD);
    k_f2bf<<<(HIDD * HIDD + 255) / 256, 256, 0, stream>>>(Wk, wWk, HIDD * HIDD);
    k_f2bf<<<(HIDD * HIDD + 255) / 256, 256, 0, stream>>>(Wv, wWv, HIDD * HIDD);

    // 1) graph start offsets from sorted batch
    k_starts_init<<<(NB + 1 + 255) / 256, 256, 0, stream>>>(starts);
    k_starts_fill<<<N_ATOMS / 256, 256, 0, stream>>>(batch, starts);

    // 2) gate GEMM + sigmoid gate -> h_gated (bf16)
    k_gate<<<N_ATOMS / 32, 256, 0, stream>>>(h_atom, c_atom, wW1, b1, W2, b2, h_gated);

    // 3) assign GEMM + softmax -> S (bf16)
    k_assign<<<N_ATOMS / 32, 256, 0, stream>>>(h_gated, wA1, ab1, A2, ab2, Sm);

    // 4) ragged per-graph segment GEMM -> cluster_raw (bf16)
    k_cluster<<<(NB * KCL) / 8, 256, 0, stream>>>(Sm, h_gated, starts, cluster_raw);

    // 5) cluster_proj + attention readout + LayerNorm -> out (fp32)
    k_readout<<<NB, 256, 0, stream>>>(cluster_raw, wWc, bc, wWk, wWv, qv, gamma, beta, outp);
}
